// InfoNCELossFull_24017457119744
// MI455X (gfx1250) — matrix operand order repaired and checked by reading, not compiled, not run on hardware
//
#include <hip/hip_runtime.h>

#define NPTS 4096
#define DIM  256
#define NB   4
#define MINIT (-3.0e38f)

typedef __attribute__((ext_vector_type(16))) __bf16 v16bf;
typedef __attribute__((ext_vector_type(8)))  __bf16 v8bf;
typedef __attribute__((ext_vector_type(8)))  float  v8f;

// ---------------- W_sym = triu(W) + triu(W).T  (diag doubled) ----------------
__global__ void k_wsym(const float* __restrict__ W, float* __restrict__ Wsym) {
    int t = blockIdx.x * blockDim.x + threadIdx.x;   // t = i*256 + j
    int i = t >> 8, j = t & 255;
    float v = 0.f;
    if (j >= i) v += W[i * DIM + j];
    if (i >= j) v += W[j * DIM + i];
    Wsym[t] = v;
}

// ---------------- fp32 -> bf16 convert ----------------
__global__ void k_cvt(const float* __restrict__ x, __bf16* __restrict__ y, int n) {
    int t = blockIdx.x * blockDim.x + threadIdx.x;
    if (t < n) y[t] = (__bf16)x[t];
}

// ---------------- AW = src_feat @ Wsym  (fp32 accum, bf16 out) ----------------
__global__ void k_aw(const float* __restrict__ src, const float* __restrict__ Wsym,
                     __bf16* __restrict__ AW) {
    int t  = blockIdx.x * blockDim.x + threadIdx.x;  // (b*N+i)*D + d
    int d  = t & (DIM - 1);
    int bi = t >> 8;
    const float* row = src + (size_t)bi * DIM;
    float acc = 0.f;
#pragma unroll 8
    for (int k = 0; k < DIM; ++k)
        acc = fmaf(row[k], Wsym[k * DIM + d], acc);
    AW[t] = (__bf16)acc;
}

// ---------------- nearest target per anchor ----------------
__global__ void k_argmin(const float* __restrict__ sx, const float* __restrict__ tx,
                         int* __restrict__ idx1, float* __restrict__ d2min) {
    int t = blockIdx.x * blockDim.x + threadIdx.x;   // b*N + i
    int b = t >> 12;
    const float* a = sx + (size_t)t * 3;
    float ax = a[0], ay = a[1], az = a[2];
    const float* p = tx + (size_t)b * NPTS * 3;
    float best = __builtin_inff();
    int   bidx = 0;
    for (int j = 0; j < NPTS; ++j) {
        float dx = ax - p[j * 3 + 0];
        float dy = ay - p[j * 3 + 1];
        float dz = az - p[j * 3 + 2];
        float d2 = dx * dx + dy * dy + dz * dz;
        if (d2 < best) { best = d2; bidx = j; }
    }
    idx1[t]  = bidx;
    d2min[t] = best;
}

__global__ void k_zero(float* acc) { acc[threadIdx.x] = 0.f; }

// ---------------- fused WMMA GEMM + mask + online logsumexp ----------------
__global__ void __launch_bounds__(256) k_lse(
    const __bf16* __restrict__ AW, const __bf16* __restrict__ TB,
    const float* __restrict__ sxyz, const float* __restrict__ txyz,
    const int* __restrict__ idx1, const float* __restrict__ d2min,
    float* __restrict__ acc)                        // [0..3]=loss sum, [4..7]=count
{
    const int lane = threadIdx.x & 31;
    const int wave = threadIdx.x >> 5;
    const int half = lane >> 4;                     // 0: lanes 0-15, 1: lanes 16-31
    const int lr   = lane & 15;
    const int b    = blockIdx.y;
    const int i0   = blockIdx.x * 128 + wave * 16;  // 16-row anchor tile per wave

    const __bf16* Arow = AW + ((size_t)(b * NPTS + i0)) * DIM;
    const __bf16* Tmat = TB + (size_t)b * NPTS * DIM;
    const float*  axyz = sxyz + ((size_t)(b * NPTS + i0)) * 3;
    const float*  pxyz = txyz + (size_t)b * NPTS * 3;

    // Preload all 8 K-chunks of the A tile (16x256 bf16) into registers.
    // 16-bit A 16x32 layout: lanes 0-15 row M=lr K{kb..kb+7, kb+16..kb+23} with kb=32kk,
    // lanes 16-31 same rows with kb=32kk+8.
    v16bf afrag[8];
#pragma unroll
    for (int kk = 0; kk < 8; ++kk) {
        int kb = kk * 32 + half * 8;
        v8bf lo = *(const v8bf*)(Arow + (size_t)lr * DIM + kb);
        v8bf hi = *(const v8bf*)(Arow + (size_t)lr * DIM + kb + 16);
        afrag[kk] = __builtin_shufflevector(lo, hi,
            0, 1, 2, 3, 4, 5, 6, 7, 8, 9, 10, 11, 12, 13, 14, 15);
    }

    float ax[8], ay[8], az[8], m[8], s[8], pos[8];
    int   pidx[8];
#pragma unroll
    for (int r = 0; r < 8; ++r) {
        int ir = r + half * 8;                      // row of C element r for this lane
        ax[r]  = axyz[ir * 3 + 0];
        ay[r]  = axyz[ir * 3 + 1];
        az[r]  = axyz[ir * 3 + 2];
        pidx[r] = idx1[b * NPTS + i0 + ir];
        m[r] = MINIT;                               // finite sentinel: keeps exp() NaN-free
        s[r] = 0.f;
        pos[r] = 0.f;
    }

    const float RN2 = 0.05f * 0.05f;
    const float NINF = -__builtin_inff();

    // -- tile j -> 8 WMMAs producing the 16x16 logits tile
    auto mma_tile = [&](int jt) -> v8f {
        // 16-bit B 32x16 layout: lanes 0-15 col N=lr K{0..7,16..23}, lanes 16-31 K{8..15,24..31}.
        const __bf16* bcol = Tmat + (size_t)(jt * 16 + lr) * DIM + half * 8;
        v8f c = {0.f, 0.f, 0.f, 0.f, 0.f, 0.f, 0.f, 0.f};
#pragma unroll
        for (int kk = 0; kk < 8; ++kk) {
            v8bf blo = *(const v8bf*)(bcol + kk * 32);
            v8bf bhi = *(const v8bf*)(bcol + kk * 32 + 16);
            v16bf bf = __builtin_shufflevector(blo, bhi,
                0, 1, 2, 3, 4, 5, 6, 7, 8, 9, 10, 11, 12, 13, 14, 15);
            c = __builtin_amdgcn_wmma_f32_16x16x32_bf16(
                    false, afrag[kk], false, bf, (short)0, c, false, false);
        }
        return c;
    };

    // -- branchless masked online-logsumexp update for one finished tile
    auto post_tile = [&](const v8f& c, int jt, float tx2, float ty2, float tz2) {
        const int j = jt * 16 + lr;
#pragma unroll
        for (int r = 0; r < 8; ++r) {
            float dx = ax[r] - tx2, dy = ay[r] - ty2, dz = az[r] - tz2;
            float d2 = dx * dx + dy * dy + dz * dz;
            bool  isPos = (j == pidx[r]);
            bool  inc   = isPos || (d2 >= RN2);     // else logit = -inf
            float v  = c[r];
            float vm = inc ? v : NINF;
            float M  = fmaxf(m[r], vm);             // finite: M >= MINIT
            s[r] = fmaf(s[r], __expf(m[r] - M), __expf(vm - M));
            m[r] = M;
            pos[r] = isPos ? v : pos[r];
        }
    };

    // software pipeline: WMMA chain for tile jt overlaps postprocess of tile jt-1.
    // unroll 2 lets the compiler ping-pong the accumulator registers instead of
    // emitting v_mov chains for cPrev = cNew.
    float txp = pxyz[lr * 3 + 0];
    float typ = pxyz[lr * 3 + 1];
    float tzp = pxyz[lr * 3 + 2];
    v8f cPrev = mma_tile(0);
#pragma unroll 2
    for (int jt = 1; jt < NPTS / 16; ++jt) {
        const int j = jt * 16 + lr;
        float nx = pxyz[j * 3 + 0];                 // prefetch next tile's xyz
        float ny = pxyz[j * 3 + 1];
        float nz = pxyz[j * 3 + 2];
        // pull B tiles a few steps ahead into the WGP cache (global_prefetch_b8)
        __builtin_prefetch(Tmat + (size_t)((((jt + 4) & 255) * 16 + lr)) * DIM + half * 16, 0, 1);
        v8f cNew = mma_tile(jt);
        post_tile(cPrev, jt - 1, txp, typ, tzp);    // independent of cNew: fills hazard slots
        cPrev = cNew;
        txp = nx; typ = ny; tzp = nz;
    }
    post_tile(cPrev, NPTS / 16 - 1, txp, typ, tzp);

    // merge partial (m,s) and pos across the 16 lanes of each half (branchless)
#pragma unroll
    for (int msk = 1; msk < 16; msk <<= 1) {
#pragma unroll
        for (int r = 0; r < 8; ++r) {
            float mo = __shfl_xor(m[r], msk, 32);
            float so = __shfl_xor(s[r], msk, 32);
            float po = __shfl_xor(pos[r], msk, 32);
            float M  = fmaxf(m[r], mo);
            s[r] = fmaf(s[r], __expf(m[r] - M), so * __expf(mo - M));
            m[r] = M;
            pos[r] += po;                           // exactly one lane contributed
        }
    }

    if (lr == 0) {
        const float RP2 = 0.1f * 0.1f;
#pragma unroll
        for (int r = 0; r < 8; ++r) {
            int ir = i0 + r + half * 8;
            if (d2min[b * NPTS + ir] < RP2) {       // anchor has a true match
                float lse = m[r] + __logf(s[r]);
                atomicAdd(&acc[b], lse - pos[r]);
                atomicAdd(&acc[4 + b], 1.f);
            }
        }
    }
}

__global__ void k_final(const float* __restrict__ acc, float* __restrict__ out) {
    float t = 0.f;
    for (int b = 0; b < NB; ++b) t += acc[b] / acc[4 + b];
    out[0] = t * (1.f / NB);
}

extern "C" void kernel_launch(void* const* d_in, const int* in_sizes, int n_in,
                              void* d_out, int out_size, void* d_ws, size_t ws_size,
                              hipStream_t stream) {
    const float* src_feat = (const float*)d_in[0];   // (4,4096,256)
    const float* tgt_feat = (const float*)d_in[1];   // (4,4096,256)
    const float* src_xyz  = (const float*)d_in[2];   // (4,4096,3)
    const float* tgt_xyz  = (const float*)d_in[3];   // (4,4096,3)
    const float* W        = (const float*)d_in[4];   // (256,256)

    char* w = (char*)d_ws;
    float*  Wsym = (float*)w;    w += (size_t)DIM * DIM * 4;        // 256 KB
    __bf16* AW   = (__bf16*)w;   w += (size_t)NB * NPTS * DIM * 2;  // 8 MB
    __bf16* TB   = (__bf16*)w;   w += (size_t)NB * NPTS * DIM * 2;  // 8 MB
    int*    idx1 = (int*)w;      w += (size_t)NB * NPTS * 4;
    float*  d2m  = (float*)w;    w += (size_t)NB * NPTS * 4;
    float*  acc  = (float*)w;    w += 64;

    k_wsym<<<DIM * DIM / 256, 256, 0, stream>>>(W, Wsym);
    k_cvt<<<(NB * NPTS * DIM) / 256, 256, 0, stream>>>(tgt_feat, TB, NB * NPTS * DIM);
    k_aw<<<(NB * NPTS * DIM) / 256, 256, 0, stream>>>(src_feat, Wsym, AW);
    k_argmin<<<(NB * NPTS) / 256, 256, 0, stream>>>(src_xyz, tgt_xyz, idx1, d2m);
    k_zero<<<1, 8, 0, stream>>>(acc);
    k_lse<<<dim3(NPTS / 128, NB), 256, 0, stream>>>(AW, TB, src_xyz, tgt_xyz, idx1, d2m, acc);
    k_final<<<1, 1, 0, stream>>>(acc, (float*)d_out);
}
